// GABlock_60593398612596
// MI455X (gfx1250) — matrix operand order repaired and checked by reading, not compiled, not run on hardware
//
#include <hip/hip_runtime.h>
#include <math.h>

// MI455X / gfx1250, wave32. All matrix math via v_wmma_f32_16x16x32_f16
// (f32 data converted to f16 at fragment load, f32 accumulation).
// All fragment strides are compile-time constants so every element load is
// base-register + immediate-offset (batched, no per-load address arithmetic,
// no xcnt/loadcnt serialization).

typedef __attribute__((ext_vector_type(16))) _Float16 v16h;
typedef __attribute__((ext_vector_type(8)))  float    v8f;

#define HW 16384   // 128*128

__device__ __forceinline__ v8f wmma_f16(v16h a, v16h b, v8f c) {
  return __builtin_amdgcn_wmma_f32_16x16x32_f16(false, a, false, b, (short)0, c,
                                                false, false);
}

// ---- WMMA fragment loaders (f32 memory -> f16 fragments) -----------------
// A (16x32, MxK) layout: lane row = lane&15; element k of VGPR pair v:
//   k = (v>>2)*16 + half*8 + (v&3)*2
// B (32x16, KxN) layout: lane col = lane&15; k = (v>>2)*16 + half*8 + (v&3)*2 (same as A)

// A, compile-time K stride KS and K limit KLIM (zero-pad above KLIM)
template <int KLIM, long KS>
__device__ __forceinline__ v16h load_a_s(const float* rp, float scale) {
  const int hf = (threadIdx.x >> 4) & 1;
  const float* rp2 = rp + (long)(hf << 3) * KS;  // fold half into base once
  float f[16];
#pragma unroll
  for (int v = 0; v < 8; ++v) {
    const int kg = (v >> 2) << 4;              // compile-time group base
    const int kc = kg + ((v & 3) << 1);        // compile-time offset part
    if (kg < KLIM) {
      f[2 * v]     = rp2[(long)kc * KS];
      f[2 * v + 1] = rp2[(long)(kc + 1) * KS];
    } else {
      f[2 * v] = 0.f;
      f[2 * v + 1] = 0.f;
    }
  }
  v16h a;
#pragma unroll
  for (int v = 0; v < 8; ++v) {
    const int kg = (v >> 2) << 4;
    const int k = kg + (hf << 3) + ((v & 3) << 1);
    a[2 * v]     = (_Float16)((kg < KLIM && k     < KLIM) ? f[2 * v]     * scale : 0.f);
    a[2 * v + 1] = (_Float16)((kg < KLIM && k + 1 < KLIM) ? f[2 * v + 1] * scale : 0.f);
  }
  return a;
}

// A, stride-1 K (weight rows, LDS rows): float2 loads, full K=32
__device__ __forceinline__ v16h load_a_w(const float* rp) {
  const int hf = (threadIdx.x >> 4) & 1;
  const float* rp2 = rp + (hf << 3);
  float2 f[8];
#pragma unroll
  for (int v = 0; v < 8; ++v)
    f[v] = *(const float2*)(rp2 + ((v >> 2) << 4) + ((v & 3) << 1));
  v16h a;
#pragma unroll
  for (int v = 0; v < 8; ++v) {
    a[2 * v]     = (_Float16)f[v].x;
    a[2 * v + 1] = (_Float16)f[v].y;
  }
  return a;
}

// B, compile-time K stride KS and K limit KLIM
template <int KLIM, long KS>
__device__ __forceinline__ v16h load_b_s(const float* cp) {
  const int hf = (threadIdx.x >> 4) & 1;
  const float* cp2 = cp + (long)(hf << 3) * KS;
  float f[16];
#pragma unroll
  for (int v = 0; v < 8; ++v) {
    const int kc = ((v >> 2) << 4) + ((v & 3) << 1);
    if (((v >> 2) << 4) < KLIM) {              // compile-time
      f[2 * v]     = cp2[(long)kc * KS];
      f[2 * v + 1] = cp2[(long)(kc + 1) * KS];
    } else {
      f[2 * v] = 0.f;
      f[2 * v + 1] = 0.f;
    }
  }
  v16h b;
#pragma unroll
  for (int v = 0; v < 8; ++v) {
    int k = ((v >> 2) << 4) + (hf << 3) + ((v & 3) << 1);
    b[2 * v]     = (_Float16)((k     < KLIM) ? f[2 * v]     : 0.f);
    b[2 * v + 1] = (_Float16)((k + 1 < KLIM) ? f[2 * v + 1] : 0.f);
  }
  return b;
}

// B, stride-1 K: float2 loads, full K=32, branchless lane-valid mask
__device__ __forceinline__ v16h load_b_s1(const float* cp, bool valid) {
  const int hf = (threadIdx.x >> 4) & 1;
  const float* cp2 = cp + (hf << 3);
  float2 f[8];
#pragma unroll
  for (int v = 0; v < 8; ++v) f[v] = *(const float2*)(cp2 + ((v >> 2) << 4) + ((v & 3) << 1));
  v16h b;
#pragma unroll
  for (int v = 0; v < 8; ++v) {
    b[2 * v]     = (_Float16)(valid ? f[v].x : 0.f);
    b[2 * v + 1] = (_Float16)(valid ? f[v].y : 0.f);
  }
  return b;
}

// ==========================================================================
// Generic conv1x1 / token-linear GEMM:  Y[M,N] = W[M,K] * X[K,N] + bias
// REQUIRES: M % 16 == 0, N % 64 == 0, K % 32 == 0 (all launches comply).
// IN_MODE : 0 = channel-major (Xb + k*LDX + n), LDX compile-time
//           1 = rolled-window gather (n = win*64+t, channel base in_ch0, roll)
// out_mode: 0 = channel-major (Yb + m*ldy + n)
//           1 = window-block  ((n/64)*96*64 + m*64 + n%64)
//           2 = grid-unshuffle into concat buffer (bz = group)
// epi     : 0 none, 1 exact GELU
// ==========================================================================
template <int IN_MODE, int LDX>
__global__ __launch_bounds__(32) void k_gemm(
    const float* __restrict__ X, const float* __restrict__ Wm,
    const float* __restrict__ bias, float* __restrict__ Y,
    int M, int N, int K, long bsx, long bsy, int ldy,
    int in_ch0, int in_shift, int out_mode, int out_ch0, int epi) {
  const int lane = threadIdx.x & 31;
  const int hf = lane >> 4;
  const int nl = lane & 15;
  const int nbase = blockIdx.x * 64;
  const int mbase = blockIdx.y * 16;
  const int bz = blockIdx.z;
  const float* Xb = X + (long)bz * bsx;

  v8f acc[4];
#pragma unroll
  for (int s4 = 0; s4 < 4; ++s4) acc[s4] = (v8f)0.0f;

  for (int k0 = 0; k0 < K; k0 += 32) {
    v16h a = load_a_w(Wm + (long)(mbase + nl) * K + k0);
#pragma unroll
    for (int s4 = 0; s4 < 4; ++s4) {
      int n = nbase + s4 * 16 + nl;
      const float* cp;
      if (IN_MODE == 0) {
        cp = Xb + (long)k0 * LDX + n;
      } else {  // rolled 8x8 window gather (stride is HW)
        int win = n >> 6, t = n & 63;
        int b2 = win >> 8, wh = (win >> 4) & 15, ww = win & 15;
        int h = ((wh << 3) + (t >> 3) + in_shift) & 127;
        int w = ((ww << 3) + (t & 7) + in_shift) & 127;
        cp = X + ((long)(b2 * 128 + in_ch0 + k0)) * HW + h * 128 + w;
      }
      acc[s4] = wmma_f16(a, load_b_s<32, (IN_MODE == 0 ? (long)LDX : (long)HW)>(cp),
                         acc[s4]);
    }
  }

#pragma unroll
  for (int s4 = 0; s4 < 4; ++s4) {
    int n = nbase + s4 * 16 + nl;
#pragma unroll
    for (int r = 0; r < 8; ++r) {
      int m = mbase + hf * 8 + r;
      float v = acc[s4][r] + bias[m];
      if (epi == 1) v = 0.5f * v * (1.0f + erff(v * 0.70710678f));
      long addr;
      if (out_mode == 0) {
        addr = (long)bz * bsy + (long)m * ldy + n;
      } else if (out_mode == 1) {
        addr = (long)(n >> 6) * (96 * 64) + (long)m * 64 + (n & 63);
      } else {  // grid unshuffle into concat (bz = group)
        int g = bz, b2 = g >> 6, ii = (g >> 3) & 7, jj = g & 7;
        int h = ((n >> 4) << 3) + ii, w = ((n & 15) << 3) + jj;
        addr = ((long)(b2 * 128 + out_ch0 + m)) * HW + h * 128 + w;
      }
      Y[addr] = v;
    }
  }
}

// ==========================================================================
// Dynamic position bias MLP: (2)->16->16->4, table (961,4)
// ==========================================================================
__global__ void k_posmlp(const float* __restrict__ w1, const float* __restrict__ b1,
                         const float* __restrict__ w2, const float* __restrict__ b2,
                         const float* __restrict__ w3, const float* __restrict__ b3,
                         float* __restrict__ ptab) {
  int e = blockIdx.x * blockDim.x + threadIdx.x;
  if (e >= 961) return;
  float rx = (float)(e / 31 - 15), ry = (float)(e % 31 - 15);
  float t[16], u[16];
#pragma unroll
  for (int i = 0; i < 16; ++i)
    t[i] = fmaxf(0.f, w1[i * 2] * rx + w1[i * 2 + 1] * ry + b1[i]);
#pragma unroll
  for (int i = 0; i < 16; ++i) {
    float s = b2[i];
#pragma unroll
    for (int j = 0; j < 16; ++j) s += w2[i * 16 + j] * t[j];
    u[i] = fmaxf(0.f, s);
  }
#pragma unroll
  for (int hh = 0; hh < 4; ++hh) {
    float s = b3[hh];
#pragma unroll
    for (int j = 0; j < 16; ++j) s += w3[hh * 16 + j] * u[j];
    ptab[e * 4 + hh] = s;
  }
}

// ==========================================================================
// Grid attention core: one wave per (query-block, head, group).
// N=256 tokens, hd=16, heads=4.  16 S-tiles in registers, half-wave shfl
// reductions (layout matches WMMA D), LDS bounce for P(D-layout)->A-layout.
// Mode 0: NCHW (B,C,HW) read through grid-shuffle map; mode 1: (g,64,256).
// ==========================================================================
template <int QM, int KM, int VM>
__global__ __launch_bounds__(32) void k_gattn(
    const float* __restrict__ Qp, int qC, int qch0,
    const float* __restrict__ Kp, int kC, int kch0,
    const float* __restrict__ Vp, int vC, int vch0,
    const float* __restrict__ ptab, float* __restrict__ Out) {
  __shared__ float scr[16][36];
  const int lane = threadIdx.x & 31;
  const int hf = lane >> 4;
  const int nl = lane & 15;
  const int qb = blockIdx.x;  // 0..15
  const int hh = blockIdx.y;  // 0..3
  const int g = blockIdx.z;   // 0..255
  const int b = g >> 6;
  const int gi = (g >> 3) & 7, gj = g & 7;   // grid-shuffle offsets
  const float scale = 0.25f;  // hd=16 ^ -0.5

  // pixel offset of token t inside image: (t>>4)*1024 + (t&15)*8 + gi*128+gj
  const int gbase = (gi << 7) + gj;

  // Q fragment (16 query tokens x hd16, K padded to 32)
  int mq = qb * 16 + nl;
  int mqp = ((mq >> 4) << 10) + ((mq & 15) << 3) + gbase;
  v16h aq;
  if (QM == 0) {
    aq = load_a_s<16, HW>(Qp + ((long)(b * qC + qch0 + hh * 16)) * HW + mqp,
                          scale);
  } else {
    aq = load_a_s<16, 256>(Qp + ((long)(g * 64 + hh * 16)) * 256 + mq, scale);
  }

  // S = Q K^T  (16 x 256)
  v8f s[16];
#pragma unroll
  for (int kb = 0; kb < 16; ++kb) {
    int tk = kb * 16 + nl;
    v8f z = (v8f)0.0f;
    if (KM == 0) {
      int tkp = ((tk >> 4) << 10) + ((tk & 15) << 3) + gbase;
      s[kb] = wmma_f16(
          aq, load_b_s<16, HW>(Kp + ((long)(b * kC + kch0 + hh * 16)) * HW + tkp),
          z);
    } else {
      s[kb] = wmma_f16(
          aq, load_b_s<16, 256>(Kp + ((long)(g * 64 + hh * 16)) * 256 + tk), z);
    }
  }

  // + dynamic position bias (D layout: row m = hf*8+r, col = kb*16+nl)
  const int tqb = qb * 16 + hf * 8;
#pragma unroll
  for (int kb = 0; kb < 16; ++kb) {
    int tk = kb * 16 + nl;
    int ik = tk >> 4, jk = tk & 15;
#pragma unroll
    for (int r = 0; r < 8; ++r) {
      int tq = tqb + r;
      int ridx = ((tq >> 4) - ik + 15) * 31 + ((tq & 15) - jk + 15);
      s[kb][r] += ptab[ridx * 4 + hh];
    }
  }

  // softmax (row reductions across 16 chunks + 16 lanes of the half-wave)
  float rmax[8], rsum[8];
#pragma unroll
  for (int r = 0; r < 8; ++r) {
    float m = -1e30f;
#pragma unroll
    for (int kb = 0; kb < 16; ++kb) m = fmaxf(m, s[kb][r]);
#pragma unroll
    for (int off = 8; off > 0; off >>= 1) m = fmaxf(m, __shfl_xor(m, off, 16));
    rmax[r] = m;
  }
#pragma unroll
  for (int r = 0; r < 8; ++r) {
    float sum = 0.f;
#pragma unroll
    for (int kb = 0; kb < 16; ++kb) {
      float e = __expf(s[kb][r] - rmax[r]);
      s[kb][r] = e;
      sum += e;
    }
#pragma unroll
    for (int off = 8; off > 0; off >>= 1) sum += __shfl_xor(sum, off, 16);
    rsum[r] = sum;
  }

  // O = P V  (accumulate over 8 key-pairs of 32)
  // VM==0: per-lane V base; element (c,vv) offset = c*2048 + (vv>>2)*1024 + (vv&3)*16 (+8)
  const float* chv = (VM == 0)
      ? Vp + ((long)(b * vC + vch0 + hh * 16 + nl)) * HW + gbase + (hf << 6)
      : Vp + ((long)(g * 64 + hh * 16 + nl)) * 256;

  v8f acc = (v8f)0.0f;
#pragma unroll
  for (int c = 0; c < 8; ++c) {
    // re-layout P (D layout) -> A layout through LDS
#pragma unroll
    for (int kbin = 0; kbin < 2; ++kbin)
#pragma unroll
      for (int r = 0; r < 8; ++r)
        scr[hf * 8 + r][kbin * 16 + nl] = s[c * 2 + kbin][r];
    asm volatile("s_wait_dscnt 0" ::: "memory");
    v16h ap = load_a_w(&scr[nl][0]);

    v16h bv;
    if (VM == 0) {
      float f[16];
#pragma unroll
      for (int vv = 0; vv < 8; ++vv) {
        f[2 * vv]     = chv[c * 2048 + ((vv >> 2) << 10) + (vv & 3) * 16];
        f[2 * vv + 1] = chv[c * 2048 + ((vv >> 2) << 10) + (vv & 3) * 16 + 8];
      }
#pragma unroll
      for (int vv = 0; vv < 8; ++vv) {
        bv[2 * vv]     = (_Float16)f[2 * vv];
        bv[2 * vv + 1] = (_Float16)f[2 * vv + 1];
      }
    } else {
      bv = load_b_s1(chv + c * 32, true);
    }
    acc = wmma_f16(ap, bv, acc);
    asm volatile("s_wait_dscnt 0" ::: "memory");
  }

  // normalize rows, store grouped (g,64,256); D col = hd channel = nl
#pragma unroll
  for (int r = 0; r < 8; ++r) {
    Out[((long)(g * 64 + hh * 16 + nl)) * 256 + (tqb + r)] = acc[r] / rsum[r];
  }
}

// ==========================================================================
// Swin / plain window attention (+ fused 32x32 projection + un-roll scatter).
// One block per window, 4 waves = 4 heads.  qkv layout: (win, 96, 64).
// ==========================================================================
__global__ __launch_bounds__(128) void k_swin(
    const float* __restrict__ qkv, const float* __restrict__ rpb,
    const float* __restrict__ pw, const float* __restrict__ pb,
    float* __restrict__ cat, int ch_out, int shift) {
  __shared__ float obuf[64][36];
  __shared__ float scr[4][16][36];
  const int lane = threadIdx.x & 31;
  const int hf = lane >> 4;
  const int nl = lane & 15;
  const int hh = threadIdx.x >> 5;  // wave = head
  const int win = blockIdx.x;
  const int bwin = win >> 8, wh = (win >> 4) & 15, ww = win & 15;
  const float scale = 0.35355339f;  // hd=8 ^ -0.5
  const long wq = (long)win * 96 * 64;

  for (int qb = 0; qb < 4; ++qb) {
    int tql = qb * 16 + nl;
    v16h aq = load_a_s<8, 64>(qkv + wq + (long)(hh * 8) * 64 + tql, scale);

    v8f s[4];
#pragma unroll
    for (int kb = 0; kb < 4; ++kb) {
      int tk = kb * 16 + nl;
      v8f z = (v8f)0.0f;
      s[kb] = wmma_f16(
          aq, load_b_s<8, 64>(qkv + wq + (long)(32 + hh * 8) * 64 + tk), z);
    }

    // relative position bias + (optional) shift mask
    const int tqb = qb * 16 + hf * 8;
#pragma unroll
    for (int kb = 0; kb < 4; ++kb) {
      int tk = kb * 16 + nl;
      int hk = tk >> 3, wk = tk & 7;
      int ridk = 0;
      if (shift) {
        int ha = wh * 8 + hk, wa = ww * 8 + wk;
        ridk = ((ha < 120) ? 0 : (ha < 124 ? 1 : 2)) * 3 +
               ((wa < 120) ? 0 : (wa < 124 ? 1 : 2));
      }
#pragma unroll
      for (int r = 0; r < 8; ++r) {
        int tq = tqb + r;
        int hq = tq >> 3, wqd = tq & 7;
        float bsv = rpb[((hq - hk + 7) * 15 + (wqd - wk + 7)) * 4 + hh];
        if (shift) {
          int ha = wh * 8 + hq, wa = ww * 8 + wqd;
          int ridq = ((ha < 120) ? 0 : (ha < 124 ? 1 : 2)) * 3 +
                     ((wa < 120) ? 0 : (wa < 124 ? 1 : 2));
          if (ridq != ridk) bsv -= 100.f;
        }
        s[kb][r] += bsv;
      }
    }

    // softmax over 64 keys
    float rmax[8], rsum[8];
#pragma unroll
    for (int r = 0; r < 8; ++r) {
      float m = -1e30f;
#pragma unroll
      for (int kb = 0; kb < 4; ++kb) m = fmaxf(m, s[kb][r]);
#pragma unroll
      for (int off = 8; off > 0; off >>= 1) m = fmaxf(m, __shfl_xor(m, off, 16));
      rmax[r] = m;
    }
#pragma unroll
    for (int r = 0; r < 8; ++r) {
      float sum = 0.f;
#pragma unroll
      for (int kb = 0; kb < 4; ++kb) {
        float e = __expf(s[kb][r] - rmax[r]);
        s[kb][r] = e;
        sum += e;
      }
#pragma unroll
      for (int off = 8; off > 0; off >>= 1) sum += __shfl_xor(sum, off, 16);
      rsum[r] = sum;
    }

    // O = P V (hd=8 -> B columns 8..15 zeroed branchlessly)
    v8f acc = (v8f)0.0f;
#pragma unroll
    for (int c = 0; c < 2; ++c) {
#pragma unroll
      for (int kbin = 0; kbin < 2; ++kbin)
#pragma unroll
        for (int r = 0; r < 8; ++r)
          scr[hh][hf * 8 + r][kbin * 16 + nl] = s[c * 2 + kbin][r];
      asm volatile("s_wait_dscnt 0" ::: "memory");
      v16h ap = load_a_w(&scr[hh][nl][0]);
      v16h bv = load_b_s1(
          qkv + wq + (long)(64 + hh * 8 + (nl & 7)) * 64 + c * 32, nl < 8);
      acc = wmma_f16(ap, bv, acc);
      asm volatile("s_wait_dscnt 0" ::: "memory");
    }
    if (nl < 8) {
#pragma unroll
      for (int r = 0; r < 8; ++r)
        obuf[tqb + r][hh * 8 + nl] = acc[r] / rsum[r];
    }
  }
  __syncthreads();

  // projection: out(32,64) = pw(32,32) @ obuf^T + pb ; 8 tiles over 4 waves
  for (int tt = 0; tt < 2; ++tt) {
    int tile = hh * 2 + tt;
    int tm = tile >> 2, tn = tile & 3;
    v16h ap = load_a_w(pw + (long)(tm * 16 + nl) * 32);
    v16h bp = load_b_s1(&obuf[tn * 16 + nl][0], true);
    v8f z = (v8f)0.0f;
    v8f o = wmma_f16(ap, bp, z);
#pragma unroll
    for (int r = 0; r < 8; ++r) {
      int m = tm * 16 + hf * 8 + r;
      int tok = tn * 16 + nl;
      int h = ((wh << 3) + (tok >> 3) + shift) & 127;
      int w = ((ww << 3) + (tok & 7) + shift) & 127;
      cat[((long)(bwin * 128 + ch_out + m)) * HW + h * 128 + w] = o[r] + pb[m];
    }
  }
}

// ==========================================================================
// out = (add1?) + channel-LayerNorm(y)*g+b + (add2?)  over (B,128,HW)
// block: 256 threads = 64 tokens x 4 channel-quarters
// ==========================================================================
__global__ __launch_bounds__(256) void k_cln(
    const float* __restrict__ y, const float* __restrict__ gam,
    const float* __restrict__ bet, const float* __restrict__ add1,
    const float* __restrict__ add2, float* __restrict__ out) {
  __shared__ float ssum[4][64], ssq[4][64], smu[64], srs[64];
  const int tl = threadIdx.x & 63, q = threadIdx.x >> 6;
  const long token = (long)blockIdx.x * 64 + tl;
  const int b = (int)(token >> 14);
  const long p = token & 16383;
  const long base = ((long)b * 128) * HW + p;

  float s = 0.f, s2 = 0.f;
  for (int c = q * 32; c < q * 32 + 32; ++c) {
    float v = y[base + (long)c * HW];
    s += v;
    s2 += v * v;
  }
  ssum[q][tl] = s;
  ssq[q][tl] = s2;
  __syncthreads();
  if (q == 0) {
    float ts = ssum[0][tl] + ssum[1][tl] + ssum[2][tl] + ssum[3][tl];
    float t2 = ssq[0][tl] + ssq[1][tl] + ssq[2][tl] + ssq[3][tl];
    float mu = ts * (1.f / 128.f);
    float var = t2 * (1.f / 128.f) - mu * mu;
    smu[tl] = mu;
    srs[tl] = rsqrtf(var + 1e-5f);
  }
  __syncthreads();
  const float mu = smu[tl], rs = srs[tl];
  for (int c = q * 32; c < q * 32 + 32; ++c) {
    long idx = base + (long)c * HW;
    float v = (y[idx] - mu) * rs * gam[c] + bet[c];
    if (add1) v += add1[idx];
    if (add2) v += add2[idx];
    out[idx] = v;
  }
}

// ==========================================================================
extern "C" void kernel_launch(void* const* d_in, const int* in_sizes, int n_in,
                              void* d_out, int out_size, void* d_ws,
                              size_t ws_size, hipStream_t stream) {
  (void)in_sizes; (void)n_in; (void)out_size; (void)ws_size;
  const float* x      = (const float*)d_in[0];
  const float* ln_g   = (const float*)d_in[1];
  const float* ln_b   = (const float*)d_in[2];
  const float* gaqkvw = (const float*)d_in[3];
  const float* gaqkvb = (const float*)d_in[4];
  const float* gagpw  = (const float*)d_in[5];
  const float* gagpb  = (const float*)d_in[6];
  const float* posw1  = (const float*)d_in[7];
  const float* posb1  = (const float*)d_in[8];
  const float* posw2  = (const float*)d_in[9];
  const float* posb2  = (const float*)d_in[10];
  const float* posw3  = (const float*)d_in[11];
  const float* posb3  = (const float*)d_in[12];
  const float* a1pw   = (const float*)d_in[13];
  const float* a1pb   = (const float*)d_in[14];
  const float* a2pw   = (const float*)d_in[15];
  const float* a2pb   = (const float*)d_in[16];
  const float* swqkvw = (const float*)d_in[17];
  const float* swqkvb = (const float*)d_in[18];
  const float* swrpb  = (const float*)d_in[19];
  const float* swpw   = (const float*)d_in[20];
  const float* swpb   = (const float*)d_in[21];
  const float* wnqkvw = (const float*)d_in[22];
  const float* wnqkvb = (const float*)d_in[23];
  const float* wnrpb  = (const float*)d_in[24];
  const float* wnpw   = (const float*)d_in[25];
  const float* wnpb   = (const float*)d_in[26];
  const float* fcw    = (const float*)d_in[27];
  const float* fcb    = (const float*)d_in[28];
  const float* n2g    = (const float*)d_in[29];
  const float* n2b    = (const float*)d_in[30];
  const float* m1w    = (const float*)d_in[31];
  const float* m1b    = (const float*)d_in[32];
  const float* m2w    = (const float*)d_in[33];
  const float* m2b    = (const float*)d_in[34];
  float* out = (float*)d_out;

  // workspace layout (floats); arena regions reused across phases
  float* wf    = (float*)d_ws;
  float* cat   = wf;                    // 4*128*HW  (concat buffer, NCHW)
  float* x1    = cat + 4l * 128 * HW;   // 4*128*HW  (post-fc residual)
  float* ptab  = x1 + 4l * 128 * HW;    // 961*4 pos-bias table
  float* arena = ptab + 4096;
  float* qkvg  = arena;                 // 4*192*HW   grid qkv (NCHW)
  float* xgrid = qkvg + 4l * 192 * HW;  // 4*64*HW
  float* t1    = xgrid + 4l * 64 * HW;  // (256,64,256)
  float* t1p   = t1 + 256l * 64 * 256;
  float* t2    = t1p + 256l * 64 * 256;
  float* swk   = arena;                 // 1024*96*64 window qkv (reuses qkvg)
  float* y1    = arena + 1024l * 96 * 64;
  float* hbuf  = y1 + 4l * 128 * HW;    // 4*256*HW MLP hidden
  float* h2    = arena;                 // 4*128*HW (reuses freed front)

  // 1. dynamic position bias table
  k_posmlp<<<dim3(4), dim3(256), 0, stream>>>(posw1, posb1, posw2, posb2,
                                              posw3, posb3, ptab);
  // 2. grid qkv + gp projections (conv1x1, WMMA GEMM)
  k_gemm<0, HW><<<dim3(256, 12, 4), dim3(32), 0, stream>>>(
      x, gaqkvw, gaqkvb, qkvg, 192, HW, 64, 128l * HW, 192l * HW, HW,
      0, 0, 0, 0, 0);
  k_gemm<0, HW><<<dim3(256, 4, 4), dim3(32), 0, stream>>>(
      x, gagpw, gagpb, xgrid, 64, HW, 64, 128l * HW, 64l * HW, HW,
      0, 0, 0, 0, 0);
  // 3. grid attention pass 1: Q=x_grid, K=k, V=v
  k_gattn<0, 0, 0><<<dim3(16, 4, 256), dim3(32), 0, stream>>>(
      xgrid, 64, 0, qkvg, 192, 64, qkvg, 192, 128, ptab, t1);
  // 4. projection a1
  k_gemm<0, 256><<<dim3(4, 4, 256), dim3(32), 0, stream>>>(
      t1, a1pw, a1pb, t1p, 64, 256, 64, 64l * 256, 64l * 256, 256,
      0, 0, 0, 0, 0);
  // 5. grid attention pass 2: Q=q, K=x_grid, V=t1p
  k_gattn<0, 0, 1><<<dim3(16, 4, 256), dim3(32), 0, stream>>>(
      qkvg, 192, 0, xgrid, 64, 0, t1p, 0, 0, ptab, t2);
  // 6. projection a2 -> concat channels 64..127 (grid unshuffle)
  k_gemm<0, 256><<<dim3(4, 4, 256), dim3(32), 0, stream>>>(
      t2, a2pw, a2pb, cat, 64, 256, 64, 64l * 256, 0, 0,
      0, 0, 2, 64, 0);
  // 7. shifted swin branch (channels 64..95 of x) -> concat ch 32..63
  k_gemm<1, HW><<<dim3(1024, 6, 1), dim3(32), 0, stream>>>(
      x, swqkvw, swqkvb, swk, 96, 4 * HW, 32, 0, 0, 0,
      64, 4, 1, 0, 0);
  k_swin<<<dim3(1024), dim3(128), 0, stream>>>(swk, swrpb, swpw, swpb, cat, 32, 4);
  // 8. plain window branch (channels 96..127 of x) -> concat ch 0..31
  k_gemm<1, HW><<<dim3(1024, 6, 1), dim3(32), 0, stream>>>(
      x, wnqkvw, wnqkvb, swk, 96, 4 * HW, 32, 0, 0, 0,
      96, 0, 1, 0, 0);
  k_swin<<<dim3(1024), dim3(128), 0, stream>>>(swk, wnrpb, wnpw, wnpb, cat, 0, 0);
  // 9. fc over concat
  k_gemm<0, HW><<<dim3(256, 8, 4), dim3(32), 0, stream>>>(
      cat, fcw, fcb, y1, 128, HW, 128, 128l * HW, 128l * HW, HW,
      0, 0, 0, 0, 0);
  // 10. x1 = shortcut + cln(y1)
  k_cln<<<dim3(1024), dim3(256), 0, stream>>>(y1, ln_g, ln_b, x, nullptr, x1);
  // 11. MLP (GELU fused in first GEMM)
  k_gemm<0, HW><<<dim3(256, 16, 4), dim3(32), 0, stream>>>(
      x1, m1w, m1b, hbuf, 256, HW, 128, 128l * HW, 256l * HW, HW,
      0, 0, 0, 0, 1);
  k_gemm<0, HW><<<dim3(256, 8, 4), dim3(32), 0, stream>>>(
      hbuf, m2w, m2b, h2, 128, HW, 256, 256l * HW, 128l * HW, HW,
      0, 0, 0, 0, 0);
  // 12. out = x1 + cln(h2) + shortcut
  k_cln<<<dim3(1024), dim3(256), 0, stream>>>(h2, n2g, n2b, x1, x, out);
}
